// NeighborGraphSAGE_BiGRU_13898514170013
// MI455X (gfx1250) — compile-verified, matrix operand order checked by reading, not run on hardware
//
#include <hip/hip_runtime.h>
#include <hip/hip_bf16.h>

// ---------------------------------------------------------------------------
// NeighborGraphSAGE + BiGRU fused kernel for MI455X (gfx1250, wave32, WMMA)
//
// B=131072, K=4, IN=64, H=64, HID=128, 3H=192, Z=576
// All GEMM stages run on v_wmma_f32_16x16x32_f16 (f16 in, f32 accumulate).
// Weights are pre-packed fp32->f16 into WMMA B-fragment order in d_ws.
// __launch_bounds__(128,4) caps VGPRs (~256) so >=4 waves/SIMD can hide the
// serial GRU dependence chain across independent row tiles.
// ---------------------------------------------------------------------------

typedef _Float16 v16h __attribute__((ext_vector_type(16)));
typedef float    v8f  __attribute__((ext_vector_type(8)));

#define B_TOT 131072
#define K_N   4
#define IN_D  64
#define H_D   64
#define HID_D 128
#define Z_D   576
#define LN_EPS 1e-5f

#define WAVES_PB 4            // 128 threads/block
#define ROWS_PW  16           // rows per wave (WMMA M)

// f16 workspace layout (element offsets). Each B-fragment tile = 512 halves:
// lane l holds 16 contiguous halves at tile*512 + l*16;  half j of lane l is
// B[k = kc*32 + (l>>4)*16 + j][n = tileN*16 + (l&15)]  = W[n][k]  (x @ W^T).
enum : int {
  WS_SELF  = 0,                 // self_W  (64x64)   -> 8 tiles
  WS_DIR0  = 4096,              // dir_W[4](64x64)   -> 4*8 tiles
  WS_FB    = 4096*5,            // fb_W    (64x64)
  WS_WIH_F = 24576,             // Wih_f   (192x64)  -> 24 tiles
  WS_WHH_F = 36864,
  WS_WIH_B = 49152,
  WS_WHH_B = 61440,
  WS_OUT   = 73728,             // out_W   (128x576) -> 144 tiles
  WS_RES   = 147456,            // res_W   (128x64)  -> 16 tiles
  WS_TOTAL = 155648             // halves  (~311 KB)
};

// --------------------------- weight pre-pack -------------------------------

__device__ __forceinline__ void pack_one(const float* W, _Float16* dst,
                                         int inDim, int elem) {
  int tile = elem >> 9;           // 512 halves per tile
  int rem  = elem & 511;
  int lane = rem >> 4;
  int j    = rem & 15;
  int kTiles = inDim >> 5;        // K-chunks of 32
  int tileN  = tile / kTiles;
  int kc     = tile % kTiles;
  int n = tileN * 16 + (lane & 15);
  int k = kc * 32 + ((lane >> 4) << 4) + j;
  dst[elem] = (_Float16)W[n * inDim + k];
}

__global__ void pack_weights(const float* selfW, const float* dirW, const float* fbW,
                             const float* wihf, const float* whhf,
                             const float* wihb, const float* whhb,
                             const float* outW, const float* resW,
                             _Float16* ws) {
  int i = blockIdx.x * blockDim.x + threadIdx.x;
  if (i >= WS_TOTAL) return;
  if (i < WS_DIR0) {
    pack_one(selfW, ws + WS_SELF, 64, i - WS_SELF);
  } else if (i < WS_FB) {
    int e = i - WS_DIR0; int d = e / 4096; e -= d * 4096;
    pack_one(dirW + d * 4096, ws + WS_DIR0 + d * 4096, 64, e);
  } else if (i < WS_WIH_F) {
    pack_one(fbW,  ws + WS_FB,    64, i - WS_FB);
  } else if (i < WS_WHH_F) {
    pack_one(wihf, ws + WS_WIH_F, 64, i - WS_WIH_F);
  } else if (i < WS_WIH_B) {
    pack_one(whhf, ws + WS_WHH_F, 64, i - WS_WHH_F);
  } else if (i < WS_WHH_B) {
    pack_one(wihb, ws + WS_WIH_B, 64, i - WS_WIH_B);
  } else if (i < WS_OUT) {
    pack_one(whhb, ws + WS_WHH_B, 64, i - WS_WHH_B);
  } else if (i < WS_RES) {
    pack_one(outW, ws + WS_OUT,  576, i - WS_OUT);
  } else {
    pack_one(resW, ws + WS_RES,   64, i - WS_RES);
  }
}

// ------------------------------ helpers ------------------------------------

__device__ __forceinline__ v8f wmma16(v16h a, v16h b, v8f c) {
  // D = A(16x32 f16) * B(32x16 f16) + C(16x16 f32)
  return __builtin_amdgcn_wmma_f32_16x16x32_f16(false, a, false, b,
                                                (short)0, c, false, false);
}

__device__ __forceinline__ v8f bcast8(float x) {
  v8f r;
#pragma unroll
  for (int i = 0; i < 8; ++i) r[i] = x;
  return r;
}

// B-fragment load from packed f16 workspace (two global_load_b128)
__device__ __forceinline__ v16h load_bfrag(const _Float16* mat, int kTiles,
                                           int tileN, int kc, int lane) {
  int tile = tileN * kTiles + kc;
  return *(const v16h*)(mat + tile * 512 + lane * 16);
}

// A-fragment straight from global fp32 (16 contiguous floats -> f16)
__device__ __forceinline__ v16h load_afrag_f32(const float* p) {
  v16h a;
  const float4* q = (const float4*)p;
#pragma unroll
  for (int i = 0; i < 4; ++i) {
    float4 f = q[i];
    a[4 * i + 0] = (_Float16)f.x;  a[4 * i + 1] = (_Float16)f.y;
    a[4 * i + 2] = (_Float16)f.z;  a[4 * i + 3] = (_Float16)f.w;
  }
  return a;
}

// A-fragment from a 16x64 f16 row-major LDS tile (K-chunk kc of 32)
__device__ __forceinline__ v16h load_afrag_lds(const _Float16* buf, int lane, int kc) {
  int row = lane & 15;
  int hi  = lane >> 4;
  return *(const v16h*)(buf + row * 64 + kc * 32 + hi * 16);
}

// Store a 16x64 fp32 C (4 N-tiles of v8f) to LDS as f16 row-major
__device__ __forceinline__ void store_c16x64(_Float16* buf, int lane, const v8f* c) {
  int n  = lane & 15;
  int hi = lane >> 4;
#pragma unroll
  for (int t = 0; t < 4; ++t)
#pragma unroll
    for (int v = 0; v < 8; ++v)
      buf[(hi * 8 + v) * 64 + t * 16 + n] = (_Float16)c[t][v];
}

__device__ __forceinline__ void lds_fence() {
  asm volatile("s_wait_dscnt 0x0" ::: "memory");
}

// One GRU direction over the 4-step neighbor sequence; streams each step's
// hidden state into the out-MLP accumulator at the right K-chunk of out_W.
__device__ __forceinline__ void run_gru(const _Float16* wih, const _Float16* whh,
                                        const float* bih, const float* bhh,
                                        _Float16* hbuf, const _Float16* xbuf,
                                        const _Float16* outw, v8f* outacc,
                                        int lane, int reverse) {
  const int n16 = lane & 15;
  v8f hreg[4];
#pragma unroll
  for (int t = 0; t < 4; ++t) hreg[t] = bcast8(0.0f);
#pragma unroll 1
  for (int i = lane; i < ROWS_PW * 64; i += 32) hbuf[i] = (_Float16)0.0f;
  lds_fence();

#pragma unroll 1
  for (int s = 0; s < 4; ++s) {
    const int ts = reverse ? (3 - s) : s;
    v16h xa[2], ha[2];
#pragma unroll
    for (int c = 0; c < 2; ++c) {
      xa[c] = load_afrag_lds(xbuf + ts * 1024, lane, c);
      ha[c] = load_afrag_lds(hbuf, lane, c);
    }
    // r,z gates: gi and gh share one accumulator (ir+hr, iz+hz)
    v8f rg[4], zg[4];
#pragma unroll
    for (int t = 0; t < 8; ++t) {
      v8f acc = bcast8(bih[t * 16 + n16] + bhh[t * 16 + n16]);
#pragma unroll
      for (int c = 0; c < 2; ++c) {
        acc = wmma16(xa[c], load_bfrag(wih, 2, t, c, lane), acc);
        acc = wmma16(ha[c], load_bfrag(whh, 2, t, c, lane), acc);
      }
#pragma unroll
      for (int v = 0; v < 8; ++v) acc[v] = 1.0f / (1.0f + __expf(-acc[v]));
      if (t < 4) rg[t] = acc; else zg[t - 4] = acc;
    }
    // n gate + state update
#pragma unroll
    for (int t = 0; t < 4; ++t) {
      v8f ai = bcast8(bih[128 + t * 16 + n16]);
      v8f ah = bcast8(bhh[128 + t * 16 + n16]);
#pragma unroll
      for (int c = 0; c < 2; ++c) {
        ai = wmma16(xa[c], load_bfrag(wih, 2, 8 + t, c, lane), ai);
        ah = wmma16(ha[c], load_bfrag(whh, 2, 8 + t, c, lane), ah);
      }
#pragma unroll
      for (int v = 0; v < 8; ++v) {
        float nn = tanhf(ai[v] + rg[t][v] * ah[v]);
        hreg[t][v] = (1.0f - zg[t][v]) * nn + zg[t][v] * hreg[t][v];
      }
    }
    // publish h for next step + transpose C->A for the out-MLP stream
    store_c16x64(hbuf, lane, hreg);
    lds_fence();
    const int kc0 = 2 + ts * 4 + (reverse ? 2 : 0);   // K-chunk of z for this h
#pragma unroll
    for (int c = 0; c < 2; ++c) {
      v16h a = load_afrag_lds(hbuf, lane, c);
#pragma unroll
      for (int t = 0; t < 8; ++t)
        outacc[t] = wmma16(a, load_bfrag(outw, 18, t, kc0 + c, lane), outacc[t]);
    }
  }
}

// ------------------------------ main kernel --------------------------------

__global__ __launch_bounds__(WAVES_PB * 32, 4)
void sage_bigru_fused(const float* __restrict__ self_feat,
                      const float* __restrict__ neigh_feat,
                      const float* __restrict__ neigh_mask,
                      const float* __restrict__ neigh_dir,
                      const float* __restrict__ self_b,
                      const float* __restrict__ dir_b,
                      const float* __restrict__ fb_b,
                      const float* __restrict__ bih_f, const float* __restrict__ bhh_f,
                      const float* __restrict__ bih_b, const float* __restrict__ bhh_b,
                      const float* __restrict__ out_b,
                      const float* __restrict__ res_b,
                      const float* __restrict__ ln_w, const float* __restrict__ ln_b,
                      const _Float16* __restrict__ ws,
                      float* __restrict__ out, int rows) {
  __shared__ __attribute__((aligned(32))) _Float16 lds_h[WAVES_PB][ROWS_PW * 64];
  __shared__ __attribute__((aligned(32))) _Float16 lds_x[WAVES_PB][K_N * ROWS_PW * 64];

  const int wid  = threadIdx.x >> 5;
  const int lane = threadIdx.x & 31;
  const int rowBase = (blockIdx.x * WAVES_PB + wid) * ROWS_PW;
  if (rowBase + ROWS_PW > rows) return;        // wave-uniform
  const int n16 = lane & 15;
  const int hi  = lane >> 4;

  __builtin_prefetch(ws, 0, 0);                // warm weight fragments
  _Float16* hbuf = lds_h[wid];
  _Float16* xbuf = lds_x[wid];

  // out-MLP accumulator [16 x 128], bias-initialized
  v8f outacc[8];
#pragma unroll
  for (int t = 0; t < 8; ++t) outacc[t] = bcast8(out_b[t * 16 + n16]);

  // ---- Phase A: h_self = relu(self @ self_W^T + b); stream into out ----
  {
    v16h aself[2];
#pragma unroll
    for (int c = 0; c < 2; ++c)
      aself[c] = load_afrag_f32(self_feat + (rowBase + n16) * IN_D + c * 32 + hi * 16);
    v8f hs[4];
#pragma unroll
    for (int t = 0; t < 4; ++t) {
      v8f acc = bcast8(self_b[t * 16 + n16]);
#pragma unroll
      for (int c = 0; c < 2; ++c)
        acc = wmma16(aself[c], load_bfrag(ws + WS_SELF, 2, t, c, lane), acc);
#pragma unroll
      for (int v = 0; v < 8; ++v) acc[v] = acc[v] > 0.0f ? acc[v] : 0.0f;
      hs[t] = acc;
    }
    store_c16x64(hbuf, lane, hs);
    lds_fence();
#pragma unroll
    for (int c = 0; c < 2; ++c) {
      v16h a = load_afrag_lds(hbuf, lane, c);
#pragma unroll
      for (int t = 0; t < 8; ++t)
        outacc[t] = wmma16(a, load_bfrag(ws + WS_OUT, 18, t, c, lane), outacc[t]);
    }
  }

  // ---- Phase B: neighbor projections (4 dir + fallback), select + mask ----
#pragma unroll 1
  for (int k = 0; k < K_N; ++k) {
    // pack per-row weight-select codes for the 8 rows this lane-half owns
    unsigned selk = 0;
#pragma unroll
    for (int v = 0; v < 8; ++v) {
      int r = rowBase + hi * 8 + v;
      float d = neigh_dir[r * K_N + k];
      float m = neigh_mask[r * K_N + k];
      int idx = (int)truncf(d * 4.0f);
      unsigned code = (m != 0.0f) ? ((idx >= 0 && idx <= 3) ? (unsigned)idx : 4u) : 7u;
      selk |= code << (4 * v);
    }
    v16h ax[2];
#pragma unroll
    for (int c = 0; c < 2; ++c)
      ax[c] = load_afrag_f32(neigh_feat + ((rowBase + n16) * K_N + k) * IN_D
                             + c * 32 + hi * 16);
    v8f xs[4];
#pragma unroll
    for (int t = 0; t < 4; ++t) xs[t] = bcast8(0.0f);
#pragma unroll 1
    for (int wsel = 0; wsel < 5; ++wsel) {
      const _Float16* Wm = (wsel < 4) ? (ws + WS_DIR0 + wsel * 4096) : (ws + WS_FB);
      const float*    bm = (wsel < 4) ? (dir_b + wsel * H_D) : fb_b;
#pragma unroll
      for (int t = 0; t < 4; ++t) {
        v8f acc = bcast8(bm[t * 16 + n16]);
#pragma unroll
        for (int c = 0; c < 2; ++c)
          acc = wmma16(ax[c], load_bfrag(Wm, 2, t, c, lane), acc);
#pragma unroll
        for (int v = 0; v < 8; ++v) {
          float rl = acc[v] > 0.0f ? acc[v] : 0.0f;
          unsigned code = (selk >> (4 * v)) & 15u;
          xs[t][v] += (code == (unsigned)wsel) ? rl : 0.0f;
        }
      }
    }
    store_c16x64(xbuf + k * 1024, lane, xs);
  }
  lds_fence();

  // ---- Phase C/D: BiGRU, streaming h_fwd_t / h_bwd_t into out-MLP ----
  run_gru(ws + WS_WIH_F, ws + WS_WHH_F, bih_f, bhh_f, hbuf, xbuf,
          ws + WS_OUT, outacc, lane, 0);
  run_gru(ws + WS_WIH_B, ws + WS_WHH_B, bih_b, bhh_b, hbuf, xbuf,
          ws + WS_OUT, outacc, lane, 1);

  // ---- Phase E: leaky-ReLU, residual GEMM, LayerNorm, store ----
#pragma unroll
  for (int t = 0; t < 8; ++t)
#pragma unroll
    for (int v = 0; v < 8; ++v) {
      float y = outacc[t][v];
      outacc[t][v] = y > 0.0f ? y : 0.2f * y;
    }
  {
    v16h ar[2];   // reload self features (L2-hot) instead of keeping them live
#pragma unroll
    for (int c = 0; c < 2; ++c)
      ar[c] = load_afrag_f32(self_feat + (rowBase + n16) * IN_D + c * 32 + hi * 16);
#pragma unroll
    for (int t = 0; t < 8; ++t) {
      v8f acc = bcast8(res_b[t * 16 + n16]);
#pragma unroll
      for (int c = 0; c < 2; ++c)
        acc = wmma16(ar[c], load_bfrag(ws + WS_RES, 2, t, c, lane), acc);
      outacc[t] += acc;
    }
  }
#pragma unroll
  for (int v = 0; v < 8; ++v) {
    float s = 0.0f, s2 = 0.0f;
#pragma unroll
    for (int t = 0; t < 8; ++t) { float y = outacc[t][v]; s += y; s2 += y * y; }
#pragma unroll
    for (int m = 1; m <= 8; m <<= 1) {      // reduce the 16 lanes of this half
      s  += __shfl_xor(s,  m, 32);
      s2 += __shfl_xor(s2, m, 32);
    }
    float mu  = s * (1.0f / 128.0f);
    float var = s2 * (1.0f / 128.0f) - mu * mu;
    float inv = rsqrtf(var + LN_EPS);
    int row = rowBase + hi * 8 + v;
#pragma unroll
    for (int t = 0; t < 8; ++t) {
      int col = t * 16 + n16;
      out[row * HID_D + col] = (outacc[t][v] - mu) * inv * ln_w[col] + ln_b[col];
    }
  }
}

// ------------------------------ launcher -----------------------------------

extern "C" void kernel_launch(void* const* d_in, const int* in_sizes, int n_in,
                              void* d_out, int out_size, void* d_ws, size_t ws_size,
                              hipStream_t stream) {
  const float* self_feat  = (const float*)d_in[0];
  const float* neigh_feat = (const float*)d_in[1];
  const float* neigh_mask = (const float*)d_in[2];
  const float* neigh_dir  = (const float*)d_in[3];
  const float* self_W = (const float*)d_in[4];
  const float* self_b = (const float*)d_in[5];
  const float* dir_W  = (const float*)d_in[6];
  const float* dir_b  = (const float*)d_in[7];
  const float* fb_W   = (const float*)d_in[8];
  const float* fb_b   = (const float*)d_in[9];
  const float* Wih_f  = (const float*)d_in[10];
  const float* Whh_f  = (const float*)d_in[11];
  const float* bih_f  = (const float*)d_in[12];
  const float* bhh_f  = (const float*)d_in[13];
  const float* Wih_b  = (const float*)d_in[14];
  const float* Whh_b  = (const float*)d_in[15];
  const float* bih_b  = (const float*)d_in[16];
  const float* bhh_b  = (const float*)d_in[17];
  const float* out_W  = (const float*)d_in[18];
  const float* out_b  = (const float*)d_in[19];
  const float* res_W  = (const float*)d_in[20];
  const float* res_b  = (const float*)d_in[21];
  const float* ln_w   = (const float*)d_in[22];
  const float* ln_b   = (const float*)d_in[23];

  _Float16* ws = (_Float16*)d_ws;
  int rows = in_sizes[0] / IN_D;

  pack_weights<<<(WS_TOTAL + 255) / 256, 256, 0, stream>>>(
      self_W, dir_W, fb_W, Wih_f, Whh_f, Wih_b, Whh_b, out_W, res_W, ws);

  int grid = (rows + WAVES_PB * ROWS_PW - 1) / (WAVES_PB * ROWS_PW);
  sage_bigru_fused<<<grid, WAVES_PB * 32, 0, stream>>>(
      self_feat, neigh_feat, neigh_mask, neigh_dir,
      self_b, dir_b, fb_b, bih_f, bhh_f, bih_b, bhh_b,
      out_b, res_b, ln_w, ln_b, ws, (float*)d_out, rows);
}
